// LowLatencyModel_26242250178806
// MI455X (gfx1250) — matrix operand order repaired and checked by reading, not compiled, not run on hardware
//
#include <hip/hip_runtime.h>
#include <math.h>

// ---------------- problem constants ----------------
#define HH 96
#define WW 192
#define PP (HH * WW)          // 18432 pixels
#define OH 384
#define OW 768
#define NCLS 19
#define OUTIMG (NCLS * OH * OW)   // 5603328

#define NTILE 96              // pixel tile per WG (96 | 192 -> never crosses a row)

typedef __attribute__((ext_vector_type(16))) _Float16 v16h;
typedef __attribute__((ext_vector_type(8)))  float    v8f;

union Frag16 { unsigned u[8]; v16h v; };

__device__ __forceinline__ unsigned pack_f16(float a, float b) {
    union { _Float16 h[2]; unsigned u; } t;
    t.h[0] = (_Float16)a; t.h[1] = (_Float16)b;
    return t.u;
}

// =====================================================================
// Halo-tiled implicit-GEMM conv via v_wmma_f32_16x16x32_f16.
//   Y[M, P] = sum_{t, c} W[M, c, t] * X[c, shifted pixel]  (+bias, *g, +beta)
// WG = 256 threads (8 waves), tile 64(M) x 96(N).
// Waves: wm = wave&3 -> 16-row M slab; wn = wave>>2 -> 48-col N slab (3 frags).
// Per 32-channel K chunk: stage X halo tile (KS x (NTILE+KS-1)) and ALL taps
// of the weight slab once; then the fully-unrolled tap loop runs KS*KS*3
// WMMAs out of LDS. K-pairs are dword-contiguous; stride-17 padding kills
// bank conflicts.
// =====================================================================
template <int KS>
__global__ __launch_bounds__(256)
void conv_wmma(const float* __restrict__ X, const float* __restrict__ Wt,
               const float* __restrict__ bias, const float* __restrict__ gam,
               const float* __restrict__ bet, float* __restrict__ Y,
               int Cin, int Cout)
{
    constexpr int TAPS = KS * KS;
    constexpr int PAD  = KS / 2;
    constexpr int XW   = NTILE + KS - 1;       // halo width
    constexpr int XCNT = KS * XW * 16;         // dwords in X tile
    constexpr int ACNT = TAPS * 64 * 16;       // dwords in A tile

    __shared__ unsigned ldsX[KS * XW * 17];    // [ky][xi][cin-pair], padded
    __shared__ unsigned ldsA[TAPS * 64 * 17];  // [tap][m-row][cin-pair], padded

    const int tid  = threadIdx.x;
    const int lane = tid & 31;
    const int wave = tid >> 5;
    const int wm   = wave & 3;                 // M sub-slab (16 rows)
    const int wn   = wave >> 2;                // N sub-slab (48 cols)
    const int n0   = blockIdx.x * NTILE;       // pixel tile base
    const int m0   = blockIdx.y * 64;          // out-channel tile base
    const int y0   = n0 / WW;
    const int x0   = n0 - y0 * WW;
    const int kc   = (Cin + 31) >> 5;

    const int half = lane >> 4;
    const int lidx = lane & 15;

    v8f acc[3] = {};

    for (int cc = 0; cc < kc; ++cc) {
        const int c0 = cc << 5;
        __syncthreads();

        // ---- stage X halo tile: KS rows x XW cols x 16 cin-pair dwords ----
        for (int i = 0; i < (XCNT + 255) / 256; ++i) {
            int idx = tid + i * 256;
            if (idx < XCNT) {
                int pr  = idx / (KS * XW);     // cin pair 0..15
                int pix = idx - pr * (KS * XW);
                int ky  = pix / XW;
                int xi  = pix - ky * XW;
                int sy = y0 + ky - PAD;
                int sx = x0 + xi - PAD;
                float v0 = 0.f, v1 = 0.f;
                if (sy >= 0 && sy < HH && sx >= 0 && sx < WW) {
                    int base = sy * WW + sx;
                    int c = c0 + pr * 2;
                    if (c     < Cin) v0 = X[(size_t)c       * PP + base];
                    if (c + 1 < Cin) v1 = X[(size_t)(c + 1) * PP + base];
                }
                ldsX[(ky * XW + xi) * 17 + pr] = pack_f16(v0, v1);
            }
        }
        // ---- stage ALL taps of the 64-row weight slab for this K chunk ----
        for (int i = 0; i < ACNT / 256; ++i) {
            int idx = tid + i * 256;           // ACNT is a multiple of 256
            int t   = idx % TAPS;
            int rp  = idx / TAPS;
            int pr  = rp & 15;
            int row = rp >> 4;
            int m = m0 + row;
            int c = c0 + pr * 2;
            float w0 = 0.f, w1 = 0.f;
            if (m < Cout) {
                if (c     < Cin) w0 = Wt[((size_t)m * Cin + c    ) * TAPS + t];
                if (c + 1 < Cin) w1 = Wt[((size_t)m * Cin + c + 1) * TAPS + t];
            }
            ldsA[(t * 64 + row) * 17 + pr] = pack_f16(w0, w1);
        }
        // speculative prefetch of next K-chunk's activations
        if (tid == 0 && (c0 + 32) < Cin)
            __builtin_prefetch(&X[(size_t)(c0 + 32) * PP + n0], 0, 1);
        __syncthreads();

        // ---- fully unrolled tap loop: KS*KS * 3 WMMAs from LDS ----
        #pragma unroll
        for (int t = 0; t < TAPS; ++t) {
            const int ky = t / KS;
            const int kx = t % KS;
            Frag16 a;
            #pragma unroll
            for (int v = 0; v < 8; ++v) {
                // A 16x32 (MxK): lane half picks K group, VGPR v -> K pair
                int apr = (v < 4) ? (half * 4 + v) : (8 + half * 4 + (v - 4));
                a.u[v] = ldsA[(t * 64 + wm * 16 + lidx) * 17 + apr];
            }
            #pragma unroll
            for (int f = 0; f < 3; ++f) {
                Frag16 b;
                int xi = wn * 48 + f * 16 + lidx + kx;   // halo-shifted col
                #pragma unroll
                for (int v = 0; v < 8; ++v)
                    b.u[v] = ldsX[(ky * XW + xi) * 17 + (v & 4) * 2 + half * 4 + (v & 3)];
                acc[f] = __builtin_amdgcn_wmma_f32_16x16x32_f16(
                             false, a.v, false, b.v, (short)0, acc[f], false, false);
            }
        }
    }

    // ---- epilogue: D 16x16 f32: lane -> N=lane&15, M=v+half*8 ----
    #pragma unroll
    for (int v = 0; v < 8; ++v) {
        int M = m0 + wm * 16 + half * 8 + v;
        if (M < Cout) {
            float bs = bias ? bias[M] : 0.f;
            float g  = gam  ? gam[M]  : 1.f;
            float bb = bet  ? bet[M]  : 0.f;
            size_t rowb = (size_t)M * PP + n0 + wn * 48 + lidx;
            #pragma unroll
            for (int f = 0; f < 3; ++f)
                Y[rowb + f * 16] = (acc[f][v] + bs) * g + bb;
        }
    }
}

// ---------------- elementwise / glue kernels ----------------
__global__ void sub_kernel(const float* __restrict__ a, const float* __restrict__ b,
                           float* __restrict__ o, int n) {
    int i = blockIdx.x * 256 + threadIdx.x;
    if (i < n) o[i] = a[i] - b[i];
}

__global__ void softmax81(float* __restrict__ K) {
    int n = blockIdx.x * 256 + threadIdx.x;
    if (n >= PP) return;
    float m = -1e30f;
    for (int k = 0; k < 81; ++k) m = fmaxf(m, K[(size_t)k * PP + n]);
    float s = 0.f;
    for (int k = 0; k < 81; ++k) {
        float e = expf(K[(size_t)k * PP + n] - m);
        K[(size_t)k * PP + n] = e;
        s += e;
    }
    float inv = 1.f / s;
    for (int k = 0; k < 81; ++k) K[(size_t)k * PP + n] *= inv;
}

// dynamic 9x9 local filtering: out[c,p] = sum_k deep[c, p shifted] * wts[k, p]
__global__ void propagate_k(const float* __restrict__ deep, const float* __restrict__ wts,
                            float* __restrict__ out) {
    int idx = blockIdx.x * 256 + threadIdx.x;   // over 256*PP
    int c = idx / PP, n = idx - c * PP;
    int y = n / WW, x = n - y * WW;
    float s = 0.f;
    const float* dp = deep + (size_t)c * PP;
    #pragma unroll 3
    for (int k = 0; k < 81; ++k) {
        int sy = y + k / 9 - 4;
        int sx = x + k % 9 - 4;
        if (sy >= 0 && sy < HH && sx >= 0 && sx < WW)
            s += dp[sy * WW + sx] * wts[(size_t)k * PP + n];
    }
    out[(size_t)c * PP + n] = s;
}

__global__ void pool_mean(const float* __restrict__ O, float* __restrict__ pooled) {
    __shared__ float red[256];
    int c = blockIdx.x, tid = threadIdx.x;
    float s = 0.f;
    for (int n = tid; n < PP; n += 256) s += O[(size_t)c * PP + n];
    red[tid] = s; __syncthreads();
    for (int st = 128; st > 0; st >>= 1) {
        if (tid < st) red[tid] += red[tid + st];
        __syncthreads();
    }
    if (tid == 0) pooled[c] = red[0] * (1.f / (float)PP);
}

__global__ void deviation_k(const float* __restrict__ pooled, const float* __restrict__ w,
                            const float* __restrict__ b, float* __restrict__ out) {
    __shared__ float red[256];
    int tid = threadIdx.x;
    red[tid] = pooled[tid] * w[tid]; __syncthreads();
    for (int st = 128; st > 0; st >>= 1) {
        if (tid < st) red[tid] += red[tid + st];
        __syncthreads();
    }
    if (tid == 0) out[0] = red[0] + b[0];
}

// bilinear 4x upsample, half-pixel centers (align_corners=False), edge clamp
__global__ void upsample4(const float* __restrict__ L, float* __restrict__ out) {
    int idx = blockIdx.x * 256 + threadIdx.x;
    if (idx >= OUTIMG) return;
    int c = idx / (OH * OW), r = idx - c * (OH * OW);
    int oy = r / OW, ox = r - oy * OW;
    float sy = (oy + 0.5f) * 0.25f - 0.5f;
    float sx = (ox + 0.5f) * 0.25f - 0.5f;
    int y0 = (int)floorf(sy); float fy = sy - (float)y0;
    int x0 = (int)floorf(sx); float fx = sx - (float)x0;
    int y0c = y0 < 0 ? 0 : (y0 > HH - 1 ? HH - 1 : y0);
    int y1c = y0 + 1 < 0 ? 0 : (y0 + 1 > HH - 1 ? HH - 1 : y0 + 1);
    int x0c = x0 < 0 ? 0 : (x0 > WW - 1 ? WW - 1 : x0);
    int x1c = x0 + 1 < 0 ? 0 : (x0 + 1 > WW - 1 ? WW - 1 : x0 + 1);
    const float* p = L + (size_t)c * PP;
    float v = (1.f - fy) * ((1.f - fx) * p[y0c * WW + x0c] + fx * p[y0c * WW + x1c])
            +        fy  * ((1.f - fx) * p[y1c * WW + x0c] + fx * p[y1c * WW + x1c]);
    out[idx] = v;
}

// ---------------- host orchestration ----------------
static inline void launch_conv(const float* X, const float* W, const float* bias,
                               const float* g, const float* bt, float* Y,
                               int Cin, int Cout, int ks, hipStream_t s) {
    dim3 grid(PP / NTILE, (Cout + 63) / 64);
    if (ks == 3)
        conv_wmma<3><<<grid, 256, 0, s>>>(X, W, bias, g, bt, Y, Cin, Cout);
    else
        conv_wmma<1><<<grid, 256, 0, s>>>(X, W, bias, g, bt, Y, Cin, Cout);
}

extern "C" void kernel_launch(void* const* d_in, const int* in_sizes, int n_in,
                              void* d_out, int out_size, void* d_ws, size_t ws_size,
                              hipStream_t stream) {
    (void)in_sizes; (void)n_in; (void)out_size; (void)ws_size;
    const float* f1_deep = (const float*)d_in[0];
    const float* f1_low  = (const float*)d_in[1];
    const float* f2_low  = (const float*)d_in[2];
    const float* wp_c1_w = (const float*)d_in[3];
    const float* wp_c1_b = (const float*)d_in[4];
    const float* wp_c2_w = (const float*)d_in[5];
    const float* wp_c2_b = (const float*)d_in[6];
    const float* wp_fc_w = (const float*)d_in[7];
    const float* wp_fc_b = (const float*)d_in[8];
    const float* sch_c1_w = (const float*)d_in[9];
    const float* sch_c2_w = (const float*)d_in[11];
    const float* sch_c2_b = (const float*)d_in[12];
    const float* sch_fc_w = (const float*)d_in[13];
    const float* sch_fc_b = (const float*)d_in[14];
    const float* ad_c1_w = (const float*)d_in[15];
    const float* ad_c1_b = (const float*)d_in[16];
    const float* ad_c2_w = (const float*)d_in[17];
    const float* ad_c2_b = (const float*)d_in[18];
    const float* ad_c3_w = (const float*)d_in[19];
    const float* ad_c3_b = (const float*)d_in[20];
    const float* rc_w = (const float*)d_in[21];
    const float* rc_b = (const float*)d_in[22];
    const float* rc_g = (const float*)d_in[23];
    const float* rc_beta = (const float*)d_in[24];
    const float* lc1_w = (const float*)d_in[25];
    const float* lc1_b = (const float*)d_in[26];
    const float* lc1_g = (const float*)d_in[27];
    const float* lc1_beta = (const float*)d_in[28];
    const float* lc2_w = (const float*)d_in[29];
    const float* lc2_b = (const float*)d_in[30];
    const float* lc2_g = (const float*)d_in[31];
    const float* lc2_beta = (const float*)d_in[32];
    const float* lc3_w = (const float*)d_in[33];
    const float* lc3_b = (const float*)d_in[34];

    float* ws = (float*)d_ws;
    float* out = (float*)d_out;

    // channel-major workspace layout (concats are pointer offsets)
    float* CAT   = ws;                         // 512 ch
    float* BUF_A = ws + (size_t)512  * PP;     // 256 ch
    float* BUF_B = ws + (size_t)768  * PP;     // 256 ch
    float* BUF_C = ws + (size_t)1024 * PP;     // 256 ch
    float* KBUF  = ws + (size_t)1280 * PP;     // 81 ch (softmax kernel)
    float* CAT2  = ws + (size_t)1361 * PP;     // 304 ch
    float* LBUF  = ws + (size_t)1665 * PP;     // 19 ch (logits)
    float* POOLD = ws + (size_t)1684 * PP;     // 256 floats

    const int ELN = 256 * PP;

    // --- WeightPredictor ---
    launch_conv(f1_low, wp_c1_w, wp_c1_b, nullptr, nullptr, CAT,             256, 256, 3, stream);
    launch_conv(f2_low, wp_c1_w, wp_c1_b, nullptr, nullptr, CAT + (size_t)256 * PP, 256, 256, 3, stream);
    launch_conv(CAT,    wp_c2_w, wp_c2_b, nullptr, nullptr, BUF_A,           512, 256, 3, stream);
    launch_conv(BUF_A,  wp_fc_w, wp_fc_b, nullptr, nullptr, KBUF,            256,  81, 1, stream);
    softmax81<<<PP / 256, 256, 0, stream>>>(KBUF);

    // --- AdaptiveScheduler (conv linearity: conv(a)-conv(b)=conv(a-b), bias cancels) ---
    sub_kernel<<<ELN / 256, 256, 0, stream>>>(f1_low, f2_low, BUF_B, ELN);
    launch_conv(BUF_B, sch_c1_w, nullptr,  nullptr, nullptr, BUF_C, 256, 256, 3, stream);
    launch_conv(BUF_C, sch_c2_w, sch_c2_b, nullptr, nullptr, BUF_B, 256, 256, 3, stream);
    pool_mean<<<256, 256, 0, stream>>>(BUF_B, POOLD);
    deviation_k<<<1, 256, 0, stream>>>(POOLD, sch_fc_w, sch_fc_b, out + OUTIMG);

    // --- Propagation: prop -> CAT[256:512]; f2_low -> CAT[0:256] ---
    propagate_k<<<ELN / 256, 256, 0, stream>>>(f1_deep, KBUF, CAT + (size_t)256 * PP);
    hipMemcpyAsync(CAT, f2_low, (size_t)ELN * sizeof(float),
                   hipMemcpyDeviceToDevice, stream);

    // --- AdaptNet ---
    launch_conv(CAT,   ad_c1_w, ad_c1_b, nullptr, nullptr, BUF_A, 512, 256, 3, stream);
    launch_conv(BUF_A, ad_c2_w, ad_c2_b, nullptr, nullptr, BUF_B, 256, 256, 3, stream);
    launch_conv(BUF_B, ad_c3_w, ad_c3_b, nullptr, nullptr, CAT2,  256, 256, 3, stream);

    // --- reduce_conv2 (1x1 + BN) -> CAT2[256:304] ---
    launch_conv(f2_low, rc_w, rc_b, rc_g, rc_beta, CAT2 + (size_t)256 * PP, 256, 48, 1, stream);

    // --- last_conv ---
    launch_conv(CAT2,  lc1_w, lc1_b, lc1_g, lc1_beta, BUF_A, 304, 256, 3, stream);
    launch_conv(BUF_A, lc2_w, lc2_b, lc2_g, lc2_beta, BUF_B, 256, 256, 3, stream);
    launch_conv(BUF_B, lc3_w, lc3_b, nullptr, nullptr, LBUF, 256, NCLS, 1, stream);

    // --- bilinear 4x upsample ---
    upsample4<<<(OUTIMG + 255) / 256, 256, 0, stream>>>(LBUF, out);
}
